// NewActivationGNN_28759101014309
// MI455X (gfx1250) — matrix operand order repaired and checked by reading, not compiled, not run on hardware
//
#include <hip/hip_runtime.h>
#include <hip/hip_bf16.h>

typedef __attribute__((ext_vector_type(16))) _Float16 v16h;
typedef __attribute__((ext_vector_type(4)))  _Float16 v4h;
typedef __attribute__((ext_vector_type(8)))  float    v8f;

#define N_NODES 50000
#define N_EDGES 800000
#define NFEAT   500
#define NHID    128
#define NCLASS  40
#define NLAYERS 4

// ---------- helpers ----------
__device__ __forceinline__ unsigned fkey(float f) {
    unsigned u = __float_as_uint(f);
    return (u & 0x80000000u) ? ~u : (u | 0x80000000u);
}
__device__ __forceinline__ float funkey(unsigned k) {
    return (k & 0x80000000u) ? __uint_as_float(k ^ 0x80000000u)
                             : __uint_as_float(~k);
}
__device__ __forceinline__ float three_seg(float v) {
    // x1=0.1, x2=0.9, c=-1:  theta*(1+e)/(1+e^theta), clamped segments
    float theta = (v - 0.1f) * (1.0f / (0.8f + 1e-8f));
    theta = fminf(fmaxf(theta, 0.0f), 1.0f);
    float seg2 = theta * 3.7182817f / (1.0f + __expf(theta));
    return (v <= 0.1f) ? 0.0f : ((v > 0.9f) ? 1.0f : seg2);
}

// =====================================================================
// Shared WMMA tile engine.
//   A: [N_NODES, K] f32 row-major, this block covers rows [m0, m0+16)
//   W: [K, NW]      f32 row-major
//   One 16x16 f32 tile per wave: wave w -> cols [16*wt, 16*wt+16) with
//   wt = min(w, NT-1 -> 0) so EVERY wave executes the WMMA with full EXEC
//   (ISA 7.12: WMMA requires EXEC all-ones; it is not skipped on EXEC==0).
//   Branchless clamped staging -> no exec churn, vectorized global loads.
// =====================================================================
template<int K, int NW, int NT, int BLOCK>
__device__ __forceinline__ v8f tile_mma(const float* __restrict__ A,
                                        const float* __restrict__ W,
                                        int m0,
                                        _Float16* lA, _Float16* lB) {
    const int tid  = threadIdx.x;
    const int wave = tid >> 5;
    const int wt   = (wave < NT) ? wave : 0;   // waves >= NT redo tile 0
    const int lane = tid & 31;
    const int m    = lane & 15;
    const int h    = lane >> 4;
    constexpr int KPAD = (K + 31) & ~31;
    v8f acc = {};
    for (int k0 = 0; k0 < KPAD; k0 += 32) {
        // ---- stage A: 16 rows x 32 cols = 128 float4 transfers ----
        for (int i = tid; i < 128; i += BLOCK) {
            const int r  = i >> 3;          // row 0..15
            const int q  = i & 7;           // float4 slot in row
            const int k4 = k0 + q * 4;
            const int kc = (k4 <= K - 4) ? k4 : (K - 4);      // clamp (K%4==0)
            const float* src = A + (size_t)(m0 + r) * K + kc;
            const float4 v = *(const float4*)src;
            if (k0 + 32 < KPAD) __builtin_prefetch(src + 32, 0, 0);
            const bool ok = (k4 < K);
            v4h hv;
            hv.x = (_Float16)(ok ? v.x : 0.0f);
            hv.y = (_Float16)(ok ? v.y : 0.0f);
            hv.z = (_Float16)(ok ? v.z : 0.0f);
            hv.w = (_Float16)(ok ? v.w : 0.0f);
            *(v4h*)(lA + r * 32 + q * 4) = hv;
        }
        // ---- stage W slab transposed: (NT*16) cols x 32 k, float4 along n ----
        for (int i = tid; i < NT * 16 * 32 / 4; i += BLOCK) {
            const int cc = i / (NT * 4);    // k offset 0..31
            const int q  = i % (NT * 4);    // float4 slot along n
            const int n4 = q * 4;
            const int k  = k0 + cc;
            const int kc = (k < K) ? k : (K - 1);
            const int nc = (n4 <= NW - 4) ? n4 : (NW - 4);    // clamp (NW%4==0)
            const float4 v = *(const float4*)(W + (size_t)kc * NW + nc);
            const bool ok = (k < K) && (n4 < NW);
            lB[(n4 + 0) * 32 + cc] = (_Float16)(ok ? v.x : 0.0f);
            lB[(n4 + 1) * 32 + cc] = (_Float16)(ok ? v.y : 0.0f);
            lB[(n4 + 2) * 32 + cc] = (_Float16)(ok ? v.z : 0.0f);
            lB[(n4 + 3) * 32 + cc] = (_Float16)(ok ? v.w : 0.0f);
        }
        __syncthreads();
        // A fragment: lane(m,h) -> K chunks [8h,8h+8) and [16+8h,16+8h+8)
        v16h a, b;
#pragma unroll
        for (int e = 0; e < 8; ++e) {
            a[e]     = lA[m * 32 + 8 * h + e];
            a[8 + e] = lA[m * 32 + 16 + 8 * h + e];
        }
        // B fragment: lane -> col n, K chunks [8h,8h+8) and [16+8h,16+8h+8)
        const int n = wt * 16 + (lane & 15);
#pragma unroll
        for (int e = 0; e < 8; ++e) {
            b[e]     = lB[n * 32 + 8 * h + e];
            b[8 + e] = lB[n * 32 + 16 + 8 * h + e];
        }
        acc = __builtin_amdgcn_wmma_f32_16x16x32_f16(false, a, false, b,
                                                     (short)0, acc, false, false);
        __syncthreads();
    }
    return acc;
}

// ---------- degree / inverse degree ----------
__global__ void deg_kernel(const int* __restrict__ er, unsigned* __restrict__ degcnt) {
    int e = blockIdx.x * blockDim.x + threadIdx.x;
    if (e < N_EDGES) atomicAdd(&degcnt[er[e]], 1u);
}
__global__ void invdeg_kernel(const unsigned* __restrict__ degcnt, float* __restrict__ invdeg) {
    int n = blockIdx.x * blockDim.x + threadIdx.x;
    if (n < N_NODES) {
        unsigned c = degcnt[n];
        invdeg[n] = 1.0f / (float)(c > 1u ? c : 1u);
    }
}

// ---------- GEMM1: h0 = features @ W_in ; x0_pre = 0.7*h0 + 0.3*relu(c); min/max ----------
__global__ __launch_bounds__(256)
void gemm_in_kernel(const float* __restrict__ A,      // [N_NODES, NFEAT]
                    const float* __restrict__ W,      // [NFEAT, NHID]
                    const float* __restrict__ cvec,   // [NHID]
                    float* __restrict__ x0,           // [N_NODES, NHID]
                    unsigned* __restrict__ minmax) {
    __shared__ __attribute__((aligned(16))) _Float16 lA[16 * 32];
    __shared__ __attribute__((aligned(16))) _Float16 lB[NHID * 32];
    __shared__ unsigned sMin, sMax;
    const int tid  = threadIdx.x;
    const int wave = tid >> 5;
    const int lane = tid & 31;
    const int m0   = blockIdx.x * 16;
    if (tid == 0) { sMin = 0xFFFFFFFFu; sMax = 0u; }
    v8f acc = tile_mma<NFEAT, NHID, 8, 256>(A, W, m0, lA, lB);

    const int h   = lane >> 4;
    const int col = wave * 16 + (lane & 15);
    const float rc = fmaxf(cvec[col], 0.0f);
    float lmin = 3.4e38f, lmax = -3.4e38f;
#pragma unroll
    for (int i = 0; i < 8; ++i) {
        int row = m0 + i + 8 * h;
        float v = 0.7f * acc[i] + 0.3f * rc;
        x0[(size_t)row * NHID + col] = v;
        lmin = fminf(lmin, v); lmax = fmaxf(lmax, v);
    }
    // wave-level butterfly reduction (avoids compiler waterfall on LDS atomics)
#pragma unroll
    for (int off = 16; off > 0; off >>= 1) {
        lmin = fminf(lmin, __shfl_xor(lmin, off, 32));
        lmax = fmaxf(lmax, __shfl_xor(lmax, off, 32));
    }
    if (lane == 0) {
        atomicMin(&sMin, fkey(lmin));
        atomicMax(&sMax, fkey(lmax));
    }
    __syncthreads();
    if (tid == 0) { atomicMin(&minmax[0], sMin); atomicMax(&minmax[1], sMax); }
}

// ---------- normalize x0 to [0,1], copy into x ----------
__global__ void norm_kernel(float* __restrict__ x0, float* __restrict__ x,
                            const unsigned* __restrict__ minmax, long long total) {
    long long i = (long long)blockIdx.x * blockDim.x + threadIdx.x;
    if (i >= total) return;
    float mn = funkey(minmax[0]);
    float mx = funkey(minmax[1]);
    float v = (x0[i] - mn) / (mx - mn + 1e-8f);
    x0[i] = v;
    x[i]  = v;
}

// ---------- GEMM2: y = x @ W  (K = NHID = 128) ----------
__global__ __launch_bounds__(256)
void gemm_hid_kernel(const float* __restrict__ A,   // [N_NODES, NHID]
                     const float* __restrict__ W,   // [NHID, NHID]
                     float* __restrict__ Y) {
    __shared__ __attribute__((aligned(16))) _Float16 lA[16 * 32];
    __shared__ __attribute__((aligned(16))) _Float16 lB[NHID * 32];
    const int m0 = blockIdx.x * 16;
    v8f acc = tile_mma<NHID, NHID, 8, 256>(A, W, m0, lA, lB);

    const int tid  = threadIdx.x;
    const int wave = tid >> 5;
    const int lane = tid & 31;
    const int h    = lane >> 4;
    const int col  = wave * 16 + (lane & 15);
#pragma unroll
    for (int i = 0; i < 8; ++i)
        Y[(size_t)(m0 + i + 8 * h) * NHID + col] = acc[i];
}

// ---------- SpMM: z[row] += invdeg[row] * y[col]  (one block per edge) ----------
__global__ __launch_bounds__(128)
void spmm_kernel(const int* __restrict__ er, const int* __restrict__ ec,
                 const float* __restrict__ invdeg,
                 const float* __restrict__ y, float* __restrict__ z) {
    const int e = blockIdx.x;
    const int f = threadIdx.x;
    const int r = er[e];
    const int c = ec[e];
    const float w = invdeg[r];
    atomicAdd(&z[(size_t)r * NHID + f], w * y[(size_t)c * NHID + f]);
}

// ---------- activation + adaptive residual ----------
__global__ void actres_kernel(const float* __restrict__ z, const float* __restrict__ x0,
                              float beta, float* __restrict__ x, long long total) {
    long long i = (long long)blockIdx.x * blockDim.x + threadIdx.x;
    if (i >= total) return;
    float v = three_seg(z[i]);
    x[i] = (1.0f - beta) * v + beta * x0[i];
}

// ---------- GEMM3: logits = x @ W_out  (40 cols padded to 48; 3 store waves) ----------
__global__ __launch_bounds__(128)
void gemm_out_kernel(const float* __restrict__ A,    // [N_NODES, NHID]
                     const float* __restrict__ W,    // [NHID, NCLASS]
                     float* __restrict__ out) {      // [N_NODES, NCLASS]
    __shared__ __attribute__((aligned(16))) _Float16 lA[16 * 32];
    __shared__ __attribute__((aligned(16))) _Float16 lB[48 * 32];
    const int m0 = blockIdx.x * 16;
    v8f acc = tile_mma<NHID, NCLASS, 3, 128>(A, W, m0, lA, lB);

    const int tid  = threadIdx.x;
    const int wave = tid >> 5;
    const int lane = tid & 31;
    const int h    = lane >> 4;
    if (wave < 3) {
        const int col = wave * 16 + (lane & 15);
        if (col < NCLASS) {
#pragma unroll
            for (int i = 0; i < 8; ++i)
                out[(size_t)(m0 + i + 8 * h) * NCLASS + col] = acc[i];
        }
    }
}

// ---------- log_softmax over 40 classes, in place on d_out ----------
__global__ void lsm_kernel(float* __restrict__ out) {
    int n = blockIdx.x * blockDim.x + threadIdx.x;
    if (n >= N_NODES) return;
    float* row = out + (size_t)n * NCLASS;
    float mx = -3.4e38f;
#pragma unroll
    for (int j = 0; j < NCLASS; ++j) mx = fmaxf(mx, row[j]);
    float s = 0.0f;
#pragma unroll
    for (int j = 0; j < NCLASS; ++j) s += __expf(row[j] - mx);
    float lse = mx + __logf(s);
#pragma unroll
    for (int j = 0; j < NCLASS; ++j) row[j] -= lse;
}

// ---------- launch ----------
extern "C" void kernel_launch(void* const* d_in, const int* in_sizes, int n_in,
                              void* d_out, int out_size, void* d_ws, size_t ws_size,
                              hipStream_t stream) {
    (void)in_sizes; (void)n_in; (void)out_size; (void)ws_size;
    const float* features = (const float*)d_in[0];
    const int*   edge_row = (const int*)d_in[1];
    const int*   edge_col = (const int*)d_in[2];
    const float* W_in     = (const float*)d_in[3];
    const float* Ws       = (const float*)d_in[4];   // [NLAYERS, NHID, NHID]
    const float* cvec     = (const float*)d_in[5];
    const float* W_out    = (const float*)d_in[6];
    float* out = (float*)d_out;

    const long long nh = (long long)N_NODES * NHID;
    char* p = (char*)d_ws;
    float*    x0     = (float*)p;    p += (size_t)nh * 4;
    float*    x      = (float*)p;    p += (size_t)nh * 4;
    float*    y      = (float*)p;    p += (size_t)nh * 4;
    float*    z      = (float*)p;    p += (size_t)nh * 4;
    unsigned* degcnt = (unsigned*)p; p += (size_t)N_NODES * 4;
    float*    invdeg = (float*)p;    p += (size_t)N_NODES * 4;
    unsigned* minmax = (unsigned*)p; p += 8;

    // reset accumulators (re-done every call -> graph-replay safe)
    hipMemsetAsync(degcnt, 0, (size_t)N_NODES * 4, stream);
    hipMemsetAsync(&minmax[0], 0xFF, 4, stream);   // min key = all ones
    hipMemsetAsync(&minmax[1], 0x00, 4, stream);   // max key = zero

    deg_kernel<<<(N_EDGES + 255) / 256, 256, 0, stream>>>(edge_row, degcnt);
    invdeg_kernel<<<(N_NODES + 255) / 256, 256, 0, stream>>>(degcnt, invdeg);

    gemm_in_kernel<<<N_NODES / 16, 256, 0, stream>>>(features, W_in, cvec, x0, minmax);
    norm_kernel<<<(unsigned)((nh + 255) / 256), 256, 0, stream>>>(x0, x, minmax, nh);

    for (int l = 0; l < NLAYERS; ++l) {
        gemm_hid_kernel<<<N_NODES / 16, 256, 0, stream>>>(x, Ws + (size_t)l * NHID * NHID, y);
        hipMemsetAsync(z, 0, (size_t)nh * 4, stream);
        spmm_kernel<<<N_EDGES, 128, 0, stream>>>(edge_row, edge_col, invdeg, y, z);
        float beta = 0.125f * (float)(l + 1);      // min(0.5,(l+1)/4*0.5)
        actres_kernel<<<(unsigned)((nh + 255) / 256), 256, 0, stream>>>(z, x0, beta, x, nh);
    }

    gemm_out_kernel<<<N_NODES / 16, 128, 0, stream>>>(x, W_out, out);
    lsm_kernel<<<(N_NODES + 255) / 256, 256, 0, stream>>>(out);
}
